// ffMoE_81767587381709
// MI455X (gfx1250) — matrix operand order repaired and checked by reading, not compiled, not run on hardware
//
#include <hip/hip_runtime.h>

// ---------------------------------------------------------------------------
// MoE FFN for MI455X (gfx1250, wave32, WMMA).
// Sparse top-2 dispatch -> grouped GEMMs on v_wmma_f32_16x16x32_bf16.
// Packed b128 LDS fills, native bf16 converts, ping-pong double buffering
// (1 barrier/K-step), global_load_async_to_lds_b128 for pass-B A tiles
// (builtin signature probed: (v4i* src, AS3 v4i* dst, i32 off, i32 cpol)),
// global_prefetch for L2-resident weight tiles.
// Workspace (~193MB): header/lists + H1 bf16 [2T][F] + out_slot f32 [2T][H].
// ---------------------------------------------------------------------------

typedef __attribute__((ext_vector_type(16))) __bf16 v16bf;
typedef __attribute__((ext_vector_type(8)))  float  v8f;
typedef __attribute__((ext_vector_type(4)))  int    v4i;
typedef __attribute__((address_space(3)))    v4i*   v4i_lds_p;

#define BM 128
#define BN 128
#define BK 32
#define LDA 40   // BK + 8 pad (bf16 units): 80B row stride, 16B-aligned rows

#if __has_builtin(__builtin_amdgcn_global_load_async_to_lds_b128)
#define ASYNC_LDS 1
#else
#define ASYNC_LDS 0
#endif

__device__ __forceinline__ void wait_async_lds() {
#if ASYNC_LDS
#if __has_builtin(__builtin_amdgcn_s_wait_asynccnt)
  __builtin_amdgcn_s_wait_asynccnt(0);
#else
  asm volatile("s_wait_asynccnt 0" ::: "memory");
#endif
#endif
}

__device__ __forceinline__ float gelu_exact(float v) {
  return 0.5f * v * (1.0f + erff(v * 0.70710678118654752f));
}

union Pack16 { __bf16 h[16]; uint4 u[2]; };
union FragU  { uint4 u[2]; v16bf v; };

__device__ __forceinline__ v16bf load_frag(const __bf16* p0, const __bf16* p1) {
  FragU f;
  f.u[0] = *(const uint4*)p0;
  f.u[1] = *(const uint4*)p1;
  return f.v;
}

// Convert 16 f32 (contiguous) -> 16 bf16, store as 2x b128 into LDS row.
__device__ __forceinline__ void cvt_store16(const float* src, __bf16* dst, bool ok) {
  Pack16 p;
  if (ok) {
    const float4* s = (const float4*)src;
    #pragma unroll
    for (int q = 0; q < 4; ++q) {
      float4 t = s[q];
      p.h[4*q+0] = (__bf16)t.x; p.h[4*q+1] = (__bf16)t.y;
      p.h[4*q+2] = (__bf16)t.z; p.h[4*q+3] = (__bf16)t.w;
    }
  } else {
    p.u[0] = make_uint4(0u,0u,0u,0u); p.u[1] = p.u[0];
  }
  ((uint4*)dst)[0] = p.u[0];
  ((uint4*)dst)[1] = p.u[1];
}

// Convert 16 f32 strided by strideN (one column of W) -> 2x b128 into LDS row.
__device__ __forceinline__ void cvt_store_col16(const float* col, size_t strideN,
                                                __bf16* dst) {
  Pack16 p;
  #pragma unroll
  for (int kk = 0; kk < 16; ++kk) p.h[kk] = (__bf16)col[(size_t)kk * strideN];
  ((uint4*)dst)[0] = p.u[0];
  ((uint4*)dst)[1] = p.u[1];
}

// --------------------------- small setup kernels ---------------------------

__global__ void init_kernel(int* hdr) {
  if (threadIdx.x < 32) hdr[threadIdx.x] = 0;   // counts[8], cursors[8], base[9]
}

__global__ __launch_bounds__(256)
void router_kernel(const float* __restrict__ x, const float* __restrict__ bbuf,
                   const float* __restrict__ Wr, const float* __restrict__ br,
                   int* __restrict__ counts, int* __restrict__ ids,
                   float* __restrict__ wts, int T, int H) {
  const int gw   = (blockIdx.x * blockDim.x + threadIdx.x) >> 5;  // one wave/token
  const int lane = threadIdx.x & 31;
  if (gw >= T) return;

  const float* xr = x + (size_t)gw * H;
  float s[8];
  #pragma unroll
  for (int e = 0; e < 8; ++e) s[e] = 0.0f;

  for (int h = lane; h < H; h += 32) {
    const float xv = xr[h];
    const float* w = Wr + (size_t)h * 8;
    #pragma unroll
    for (int e = 0; e < 8; ++e) s[e] += xv * w[e];
  }
  #pragma unroll
  for (int off = 16; off >= 1; off >>= 1) {
    #pragma unroll
    for (int e = 0; e < 8; ++e) s[e] += __shfl_xor(s[e], off, 32);
  }
  if (lane != 0) return;

  float sc[8], mx = -1e30f;
  #pragma unroll
  for (int e = 0; e < 8; ++e) { sc[e] = s[e] + br[e]; mx = fmaxf(mx, sc[e]); }
  float p[8], sum = 0.0f;
  #pragma unroll
  for (int e = 0; e < 8; ++e) { p[e] = __expf(sc[e] - mx); sum += p[e]; }
  const float inv = 1.0f / sum;
  #pragma unroll
  for (int e = 0; e < 8; ++e) p[e] *= inv;

  float sel[8];                                   // biased selection
  #pragma unroll
  for (int e = 0; e < 8; ++e) sel[e] = p[e] + bbuf[e];
  int e0 = 0;
  #pragma unroll
  for (int e = 1; e < 8; ++e) if (sel[e] > sel[e0]) e0 = e;
  int e1 = (e0 == 0) ? 1 : 0;
  #pragma unroll
  for (int e = 0; e < 8; ++e) if (e != e0 && sel[e] > sel[e1]) e1 = e;

  int i0 = 0;                                     // unbiased top-2 probs
  #pragma unroll
  for (int e = 1; e < 8; ++e) if (p[e] > p[i0]) i0 = e;
  int i1 = (i0 == 0) ? 1 : 0;
  #pragma unroll
  for (int e = 0; e < 8; ++e) if (e != i0 && p[e] > p[i1]) i1 = e;
  const float p0 = p[i0], p1 = p[i1];
  const float wn = 1.0f / (p0 + p1);

  ids[2 * gw]     = e0;
  ids[2 * gw + 1] = e1;
  wts[2 * gw]     = p0 * wn;
  wts[2 * gw + 1] = p1 * wn;
  atomicAdd(&counts[e0], 1);
  atomicAdd(&counts[e1], 1);
}

__global__ void prefix_kernel(const int* __restrict__ counts, int* __restrict__ base, int E) {
  if (threadIdx.x == 0) {
    int acc = 0;
    for (int e = 0; e < E; ++e) { base[e] = acc; acc += counts[e]; }
    base[E] = acc;
  }
}

__global__ void dispatch_kernel(const int* __restrict__ ids, const int* __restrict__ base,
                                int* __restrict__ cursors, int* __restrict__ slot_tok,
                                int* __restrict__ slot_map, int T) {
  int i = blockIdx.x * blockDim.x + threadIdx.x;
  if (i >= 2 * T) return;
  const int e   = ids[i];
  const int pos = atomicAdd(&cursors[e], 1);
  const int s   = base[e] + pos;
  slot_tok[s]   = i >> 1;
  slot_map[i]   = s;
}

// ------------------------- GEMM pass A: H1 = GELU(Xg*W1+b1) ----------------

__global__ __launch_bounds__(256)
void moe_h1_kernel(const float* __restrict__ x, const float* __restrict__ W1,
                   const float* __restrict__ b1, const int* __restrict__ basep,
                   const int* __restrict__ slot_tok, __bf16* __restrict__ H1,
                   int H, int F) {
  const int e       = blockIdx.z;
  const int slotBeg = basep[e], slotEnd = basep[e + 1];
  const int m0      = blockIdx.y * BM;
  if (m0 >= slotEnd - slotBeg) return;                 // uniform early-exit
  const int baseSlot = slotBeg + m0;
  const int n0       = blockIdx.x * BN;

  __shared__ __bf16 lA[2][BM][LDA];
  __shared__ __bf16 lB[2][BN][LDA];

  const int tid = threadIdx.x;
  const int lane = tid & 31, wid = tid >> 5;
  const int wM = (wid & 1) * 64;
  const int wN = (wid >> 1) * 32;

  // A fill: 2 threads/row, 16 f32 each (gather + cvt)
  const int ar = tid >> 1;
  const int ac = (tid & 1) * 16;
  const int aslot = baseSlot + ar;
  const float* arow = nullptr;
  if (aslot < slotEnd) arow = x + (size_t)slot_tok[aslot] * H + ac;

  // B fill: thread owns one column n (coalesced across lanes), 16 K values
  const int bn  = tid & 127;
  const int bkh = (tid >> 7) * 16;
  const float* Wb = W1 + (size_t)e * H * F + n0 + bn;   // column base

  v8f acc[4][2] = {};
  const int nk = H / BK;

  // prologue fill buffer 0
  cvt_store16(arow, &lA[0][ar][ac], arow != nullptr);
  cvt_store_col16(Wb + (size_t)bkh * F, F, &lB[0][bn][bkh]);
  __syncthreads();

  for (int kt = 0; kt < nk; ++kt) {
    const int cur = kt & 1;
    const int k0n = (kt + 1) * BK;

    // compute on cur
    const int mr = lane & 15;
    const int gA = (lane >> 4) * 8;
    const int gB = (lane >> 4) * 8;
    v16bf afr[4], bfr[2];
    #pragma unroll
    for (int i = 0; i < 4; ++i) {
      const __bf16* rp = &lA[cur][wM + i * 16 + mr][0];
      afr[i] = load_frag(rp + gA, rp + gA + 16);
    }
    #pragma unroll
    for (int j = 0; j < 2; ++j) {
      const __bf16* rp = &lB[cur][wN + j * 16 + mr][0];
      bfr[j] = load_frag(rp + gB, rp + gB + 16);
    }
    #pragma unroll
    for (int i = 0; i < 4; ++i) {
      #pragma unroll
      for (int j = 0; j < 2; ++j) {
        acc[i][j] = __builtin_amdgcn_wmma_f32_16x16x32_bf16(
            false, afr[i], false, bfr[j], (short)0, acc[i][j], false, false);
      }
    }

    // fill nxt while cur computes
    if (kt + 1 < nk) {
      const int nxt = cur ^ 1;
      cvt_store16(arow ? (arow + k0n) : nullptr, &lA[nxt][ar][ac], arow != nullptr);
      cvt_store_col16(Wb + (size_t)(k0n + bkh) * F, F, &lB[nxt][bn][bkh]);
      if (kt + 2 < nk)   // weights are L2-resident (192MB L2); hide latency
        __builtin_prefetch(Wb + (size_t)(k0n + BK + bkh) * F, 0, 3);
    }
    __syncthreads();
  }

  // Epilogue: +b1, exact GELU, store bf16 (C: VGPR r -> M = r + 8*(lane>=16))
  const int nl = lane & 15;
  const int mh = (lane >> 4) * 8;
  #pragma unroll
  for (int j = 0; j < 2; ++j) {
    const int gn = n0 + wN + j * 16 + nl;
    const float bias = b1[(size_t)e * F + gn];
    #pragma unroll
    for (int i = 0; i < 4; ++i) {
      #pragma unroll
      for (int r = 0; r < 8; ++r) {
        const int slot = baseSlot + wM + i * 16 + mh + r;
        if (slot < slotEnd) {
          H1[(size_t)slot * F + gn] = (__bf16)gelu_exact(acc[i][j][r] + bias);
        }
      }
    }
  }
}

// ------------------------- GEMM pass B: out_slot = H1*W2+b2 ----------------

__global__ __launch_bounds__(256)
void moe_out_kernel(const __bf16* __restrict__ H1, const float* __restrict__ W2,
                    const float* __restrict__ b2, const int* __restrict__ basep,
                    float* __restrict__ outSlot, int F, int H, int totalSlots) {
  const int e       = blockIdx.z;
  const int slotBeg = basep[e], slotEnd = basep[e + 1];
  const int m0      = blockIdx.y * BM;
  if (m0 >= slotEnd - slotBeg) return;
  const int baseSlot = slotBeg + m0;
  const int n0       = blockIdx.x * BN;

  __shared__ __bf16 lA[2][BM][LDA];
  __shared__ __bf16 lB[2][BN][LDA];

  const int tid = threadIdx.x;
  const int lane = tid & 31, wid = tid >> 5;
  const int wM = (wid & 1) * 64;
  const int wN = (wid >> 1) * 32;

  const int ar = tid >> 1;
  const int ac = (tid & 1) * 16;
  const int aslot = baseSlot + ar;
  const bool aok  = (aslot < totalSlots);
  const __bf16* agp = H1 + (size_t)aslot * F + ac;     // bf16, no conversion

  const int bn  = tid & 127;
  const int bkh = (tid >> 7) * 16;
  const float* Wb = W2 + (size_t)e * F * H + n0 + bn;

  v8f acc[4][2] = {};
  const int nk = F / BK;

  // A-tile copy helper: bf16 -> LDS (async CDNA5 path when available)
  auto copyA = [&](int k0, int buf) {
    __bf16* ld = &lA[buf][ar][ac];
    if (aok) {
#if ASYNC_LDS
      __builtin_amdgcn_global_load_async_to_lds_b128(
          (v4i*)(agp + k0), (v4i_lds_p)(ld), 0, 0);
      __builtin_amdgcn_global_load_async_to_lds_b128(
          (v4i*)(agp + k0 + 8), (v4i_lds_p)(ld + 8), 0, 0);
#else
      const uint4* g = (const uint4*)(agp + k0);
      ((uint4*)ld)[0] = g[0];
      ((uint4*)ld)[1] = g[1];
#endif
    } else {
      uint4 z = make_uint4(0u,0u,0u,0u);
      ((uint4*)ld)[0] = z;
      ((uint4*)ld)[1] = z;
    }
  };

  copyA(0, 0);
  cvt_store_col16(Wb + (size_t)bkh * H, H, &lB[0][bn][bkh]);
  wait_async_lds();
  __syncthreads();

  for (int kt = 0; kt < nk; ++kt) {
    const int cur = kt & 1;
    const int k0n = (kt + 1) * BK;

    const int mr = lane & 15;
    const int gA = (lane >> 4) * 8;
    const int gB = (lane >> 4) * 8;
    v16bf afr[4], bfr[2];
    #pragma unroll
    for (int i = 0; i < 4; ++i) {
      const __bf16* rp = &lA[cur][wM + i * 16 + mr][0];
      afr[i] = load_frag(rp + gA, rp + gA + 16);
    }
    #pragma unroll
    for (int j = 0; j < 2; ++j) {
      const __bf16* rp = &lB[cur][wN + j * 16 + mr][0];
      bfr[j] = load_frag(rp + gB, rp + gB + 16);
    }
    #pragma unroll
    for (int i = 0; i < 4; ++i) {
      #pragma unroll
      for (int j = 0; j < 2; ++j) {
        acc[i][j] = __builtin_amdgcn_wmma_f32_16x16x32_bf16(
            false, afr[i], false, bfr[j], (short)0, acc[i][j], false, false);
      }
    }

    if (kt + 1 < nk) {
      const int nxt = cur ^ 1;
      copyA(k0n, nxt);
      cvt_store_col16(Wb + (size_t)(k0n + bkh) * H, H, &lB[nxt][bn][bkh]);
      if (kt + 2 < nk)
        __builtin_prefetch(Wb + (size_t)(k0n + BK + bkh) * H, 0, 3);
    }
    wait_async_lds();
    __syncthreads();
  }

  const int nl = lane & 15;
  const int mh = (lane >> 4) * 8;
  #pragma unroll
  for (int j = 0; j < 2; ++j) {
    const int gn = n0 + wN + j * 16 + nl;
    const float bias = b2[(size_t)e * H + gn];
    #pragma unroll
    for (int i = 0; i < 4; ++i) {
      #pragma unroll
      for (int r = 0; r < 8; ++r) {
        const int slot = baseSlot + wM + i * 16 + mh + r;
        if (slot < slotEnd) {
          outSlot[(size_t)slot * H + gn] = acc[i][j][r] + bias;
        }
      }
    }
  }
}

// --------------------------- weighted combine ------------------------------

__global__ void combine_kernel(const float* __restrict__ outSlot,
                               const int* __restrict__ slot_map,
                               const float* __restrict__ wts,
                               float* __restrict__ out, int T, int H) {
  const int per = H >> 2;
  int i = blockIdx.x * blockDim.x + threadIdx.x;
  if (i >= T * per) return;
  const int t = i / per;
  const int c = i - t * per;
  const int s0 = slot_map[2 * t], s1 = slot_map[2 * t + 1];
  const float w0 = wts[2 * t], w1 = wts[2 * t + 1];
  const float4 a = ((const float4*)(outSlot + (size_t)s0 * H))[c];
  const float4 b = ((const float4*)(outSlot + (size_t)s1 * H))[c];
  float4 r;
  r.x = w0 * a.x + w1 * b.x;
  r.y = w0 * a.y + w1 * b.y;
  r.z = w0 * a.z + w1 * b.z;
  r.w = w0 * a.w + w1 * b.w;
  ((float4*)out)[i] = r;
}

// ------------------------------- launcher ----------------------------------

extern "C" void kernel_launch(void* const* d_in, const int* in_sizes, int n_in,
                              void* d_out, int out_size, void* d_ws, size_t ws_size,
                              hipStream_t stream) {
  const float* x    = (const float*)d_in[0];
  const float* bbuf = (const float*)d_in[1];
  const float* Wr   = (const float*)d_in[2];
  const float* br   = (const float*)d_in[3];
  const float* W1   = (const float*)d_in[4];
  const float* b1   = (const float*)d_in[5];
  const float* W2   = (const float*)d_in[6];
  const float* b2   = (const float*)d_in[7];

  const int E = in_sizes[1];            // 8
  const int H = in_sizes[2] / E;        // 1024
  const int F = in_sizes[5] / E;        // 4096
  const int T = in_sizes[0] / H;        // 8192 tokens
  const int totalSlots = 2 * T;         // k = 2

  char* ws = (char*)d_ws;
  int*   counts   = (int*)ws;
  int*   cursors  = counts + 8;
  int*   basep    = cursors + 8;
  int*   ids      = (int*)(ws + 256);
  float* wts      = (float*)(ws + 256 + (size_t)totalSlots * 4);
  int*   slot_tok = (int*)(ws + 256 + (size_t)totalSlots * 8);
  int*   slot_map = (int*)(ws + 256 + (size_t)totalSlots * 12);
  size_t off = 256 + (size_t)totalSlots * 16;
  off = (off + 255) & ~(size_t)255;
  __bf16* H1 = (__bf16*)(ws + off);
  off += (size_t)totalSlots * F * 2;
  float* outSlot = (float*)(ws + off);

  init_kernel<<<1, 32, 0, stream>>>(counts);
  router_kernel<<<(T * 32 + 255) / 256, 256, 0, stream>>>(x, bbuf, Wr, br,
                                                          counts, ids, wts, T, H);
  prefix_kernel<<<1, 1, 0, stream>>>(counts, basep, E);
  dispatch_kernel<<<(2 * T + 255) / 256, 256, 0, stream>>>(ids, basep, cursors,
                                                           slot_tok, slot_map, T);

  dim3 gA(F / BN, (totalSlots + BM - 1) / BM, E);
  moe_h1_kernel<<<gA, 256, 0, stream>>>(x, W1, b1, basep, slot_tok, H1, H, F);

  dim3 gB(H / BN, (totalSlots + BM - 1) / BM, E);
  moe_out_kernel<<<gB, 256, 0, stream>>>(H1, W2, b2, basep, outSlot, F, H, totalSlots);

  combine_kernel<<<(T * (H / 4) + 255) / 256, 256, 0, stream>>>(outSlot, slot_map,
                                                                wts, (float*)d_out, T, H);
}
